// PointNet_78675210928148
// MI455X (gfx1250) — matrix operand order repaired and checked by reading, not compiled, not run on hardware
//
#include <hip/hip_runtime.h>
#include <hip/hip_bf16.h>

typedef __attribute__((ext_vector_type(16))) _Float16 v16h;
typedef __attribute__((ext_vector_type(8)))  float    v8f;

// ---------------------------------------------------------------------------
// pos L1-normalize: pn[i] = pos[i] / max(|x|+|y|+|z|, 1e-12); also f16 copy
// ---------------------------------------------------------------------------
__global__ void normalize_kernel(const float* __restrict__ pos,
                                 float* __restrict__ pn,
                                 _Float16* __restrict__ pn16, int N) {
  int i = blockIdx.x * blockDim.x + threadIdx.x;
  if (i >= N) return;
  float x = pos[3*i+0], y = pos[3*i+1], z = pos[3*i+2];
  float s = fabsf(x) + fabsf(y) + fabsf(z);
  s = fmaxf(s, 1e-12f);
  float inv = 1.0f / s;
  x *= inv; y *= inv; z *= inv;
  pn[3*i+0] = x; pn[3*i+1] = y; pn[3*i+2] = z;
  pn16[3*i+0] = (_Float16)x; pn16[3*i+1] = (_Float16)y; pn16[3*i+2] = (_Float16)z;
}

// ---------------------------------------------------------------------------
// zero fill (atomic-max identity; reference fills empty segments with 0)
// ---------------------------------------------------------------------------
__global__ void zero_kernel(float* __restrict__ p, int n) {
  for (int i = blockIdx.x * blockDim.x + threadIdx.x; i < n;
       i += gridDim.x * blockDim.x)
    p[i] = 0.0f;
}

// ---------------------------------------------------------------------------
// f32 -> f16 elementwise (activation repack, once per node instead of per edge)
// ---------------------------------------------------------------------------
__global__ void cvt_kernel(const float* __restrict__ src,
                           _Float16* __restrict__ dst, int n) {
  for (int i = blockIdx.x * blockDim.x + threadIdx.x; i < n;
       i += gridDim.x * blockDim.x)
    dst[i] = (_Float16)src[i];
}

// ---------------------------------------------------------------------------
// Edge message-passing layer:
//   x_e = [h16[src] (C_H), pn[src]-pn[dst] (3)]      (K1 = C_H+3 features)
//   m_e = relu(x_e @ Wa + ba) @ Wb + bb              (C_OUT)
//   out[dst] = max(0, max_e m_e)   via uint atomicMax on float bits (>=0)
//
// One wave per block, tiles of 16 edges, WMMA f16 -> f32 (16x16x32).
// GEMM1 bias is folded into the GEMM: A slot k==K1 holds 1.0, weight row K1
// holds ba -> C operand is inline zero (no accumulator init movs).
// A layout (16-bit 16x32): lane L: M = L&15, half p = L>>4;
//   element i -> K = ks*32 + (i<8?0:16) + p*8 + (i&7)   (two 8-contig runs)
// B layout (16-bit 32x16): lane L: N = L&15; element i -> K = ks*32+(i<8?0:16)+(L>>4)*8+(i&7) (same as A)
// C/D layout (f32 16x16): lane L: N = L&15; element r -> M = r + 8*(L>>4)
// ---------------------------------------------------------------------------
template<int C_H, int C_OUT>
__global__ __launch_bounds__(32)
void edge_layer_kernel(const _Float16* __restrict__ h16,
                       const float* __restrict__ pn,
                       const int*   __restrict__ face,
                       const float* __restrict__ Wa, const float* __restrict__ ba,
                       const float* __restrict__ Wb, const float* __restrict__ bb,
                       float* __restrict__ out, int N, int F) {
  constexpr int K1  = C_H + 3;
  constexpr int KS1 = (K1 + 3 + 31) / 32;  // K-steps of GEMM1 (+1 slot for bias)
  constexpr int KS2 = C_OUT / 32;          // K-steps of GEMM2
  constexpr int NT  = C_OUT / 16;          // N tiles

  __shared__ _Float16 Y[16 * C_OUT];       // intermediate tile, D->A re-stripe
  __shared__ int sdst[16];

  const int lane = threadIdx.x;
  const int m    = lane & 15;              // A-row (edge slot) / B,C,D column
  const int p    = lane >> 4;

  // ---- preload weights (B layout) + biases, once per wave ----
  v16h wAreg[KS1][NT];
  v16h wBreg[KS2][NT];
  float biasB[NT];
#pragma unroll
  for (int t = 0; t < NT; ++t) {
    const int n = m + 16 * t;
    biasB[t] = bb[n];
#pragma unroll
    for (int ks = 0; ks < KS1; ++ks)
#pragma unroll
      for (int i = 0; i < 16; ++i) {
        const int k = ks * 32 + ((i < 8) ? 0 : 16) + p * 8 + (i & 7);
        _Float16 w;
        if (k < K1)       w = (_Float16)Wa[k * C_OUT + n];
        else if (k == K1) w = (_Float16)ba[n];           // bias row
        else              w = (_Float16)0.0f;
        wAreg[ks][t][i] = w;
      }
#pragma unroll
    for (int ks = 0; ks < KS2; ++ks)
#pragma unroll
      for (int i = 0; i < 16; ++i) {
        const int k = ks * 32 + ((i < 8) ? 0 : 16) + p * 8 + (i & 7);
        wBreg[ks][t][i] = (_Float16)Wb[k * C_OUT + n];
      }
  }

  // grid-stride over (segment s in 0..5, j-tile) -- per-lane index needs no div
  const int tps    = (F + 15) >> 4;   // j-tiles per segment
  const int ntiles = 6 * tps;

  for (int tile = blockIdx.x; tile < ntiles; tile += gridDim.x) {
    __syncthreads();  // WAR on Y/sdst vs previous iteration (wave32 block: cheap)

    // ---- decode edge; s is wave-uniform -> SALU ----
    const int s    = tile / tps;
    const int jt   = tile - s * tps;
    const int j0   = jt * 16 + m;
    const bool ok  = (j0 < F);
    const int j    = ok ? j0 : (F - 1);
    const int srow = s >> 1;
    const int drow = (int)((0x100221u >> (4 * s)) & 0xFu);  // {1,2,2,0,0,1}
    const int src  = face[srow * F + j];
    const int dst  = face[drow * F + j];
    if (p == 0) sdst[m] = ok ? dst : -1;

    const _Float16 d0 = (_Float16)(pn[3*src+0] - pn[3*dst+0]);
    const _Float16 d1 = (_Float16)(pn[3*src+1] - pn[3*dst+1]);
    const _Float16 d2 = (_Float16)(pn[3*src+2] - pn[3*dst+2]);

    // ---- gather features straight into A layout (8-half contiguous runs) ----
    const _Float16* __restrict__ hrow = h16 + (long long)src * C_H;
    v16h a1[KS1];
#pragma unroll
    for (int ks = 0; ks < KS1; ++ks)
#pragma unroll
      for (int i = 0; i < 16; ++i) {
        const int k = ks * 32 + ((i < 8) ? 0 : 16) + p * 8 + (i & 7);
        _Float16 v;
        if (k < C_H)           v = hrow[k];
        else if (k == C_H)     v = d0;
        else if (k == C_H + 1) v = d1;
        else if (k == C_H + 2) v = d2;
        else if (k == K1)      v = (_Float16)1.0f;  // bias slot
        else                   v = (_Float16)0.0f;
        a1[ks][i] = v;
      }

    // ---- GEMM1: y = relu(x @ Wa + ba)  (C starts as inline zero) ----
    v8f acc[NT];
#pragma unroll
    for (int t = 0; t < NT; ++t) {
      v8f c = {};
#pragma unroll
      for (int ks = 0; ks < KS1; ++ks)
        c = __builtin_amdgcn_wmma_f32_16x16x32_f16(
            false, a1[ks], false, wAreg[ks][t], (short)0, c, false, false);
      acc[t] = c;
    }

    // ---- D layout -> LDS (with ReLU) ----
#pragma unroll
    for (int t = 0; t < NT; ++t)
#pragma unroll
      for (int r = 0; r < 8; ++r)
        Y[(r + 8 * p) * C_OUT + (m + 16 * t)] = (_Float16)fmaxf(acc[t][r], 0.0f);
    __syncthreads();

    // ---- LDS -> A layout for GEMM2 (contiguous 8-half runs -> ds b128) ----
    v16h a2[KS2];
#pragma unroll
    for (int ks = 0; ks < KS2; ++ks)
#pragma unroll
      for (int i = 0; i < 16; ++i) {
        const int k = ks * 32 + ((i < 8) ? 0 : 16) + p * 8 + (i & 7);
        a2[ks][i] = Y[m * C_OUT + k];
      }

    // ---- GEMM2: m = y @ Wb + bb ----
    v8f acc2[NT];
#pragma unroll
    for (int t = 0; t < NT; ++t) {
      v8f c;
#pragma unroll
      for (int r = 0; r < 8; ++r) c[r] = biasB[t];
#pragma unroll
      for (int ks = 0; ks < KS2; ++ks)
        c = __builtin_amdgcn_wmma_f32_16x16x32_f16(
            false, a2[ks], false, wBreg[ks][t], (short)0, c, false, false);
      acc2[t] = c;
    }

    // ---- clamp >= 0, scatter atomic-max (float bits as uint: monotone >=0) ----
#pragma unroll
    for (int r = 0; r < 8; ++r) {
      const int row = r + 8 * p;
      const int d   = sdst[row];
      if (d >= 0) {
        float* __restrict__ orow = out + (long long)d * C_OUT;
#pragma unroll
        for (int t = 0; t < NT; ++t) {
          const float v = fmaxf(acc2[t][r], 0.0f);
          atomicMax((unsigned int*)(orow + m + 16 * t), __float_as_uint(v));
        }
      }
    }
  }
}

// ---------------------------------------------------------------------------
// global_max_pool over sorted batch ids: g[64][64], values >= 0, g zero-init
// ---------------------------------------------------------------------------
__global__ __launch_bounds__(256)
void pool_kernel(const float* __restrict__ h, const int* __restrict__ batch,
                 float* __restrict__ g, int N) {
  __shared__ int sb[256];
  const int base = blockIdx.x * 256;
  const int t = threadIdx.x;
  {
    const int i = base + t;
    sb[t] = (i < N) ? batch[i] : -1;
  }
  __syncthreads();
  const int c   = t & 63;
  const int sub = t >> 6;  // 0..3
  int curg = -1;
  float vmax = 0.0f;
  for (int k = sub; k < 256; k += 4) {
    const int i = base + k;
    if (i >= N) break;
    const int gb = sb[k];
    if (gb != curg) {
      if (curg >= 0)
        atomicMax((unsigned int*)(g + curg * 64 + c), __float_as_uint(vmax));
      curg = gb;
      vmax = 0.0f;
    }
    vmax = fmaxf(vmax, h[(long long)i * 64 + c]);
  }
  if (curg >= 0)
    atomicMax((unsigned int*)(g + curg * 64 + c), __float_as_uint(vmax));
}

// ---------------------------------------------------------------------------
// classifier: out[64,41] = g[64,64] @ Wc[64,41] + bc   (tiny, full f32)
// ---------------------------------------------------------------------------
__global__ void classifier_kernel(const float* __restrict__ g,
                                  const float* __restrict__ Wc,
                                  const float* __restrict__ bc,
                                  float* __restrict__ out) {
  const int idx = blockIdx.x * blockDim.x + threadIdx.x;
  if (idx >= 64 * 41) return;
  const int gi = idx / 41, o = idx - gi * 41;
  float acc = bc[o];
#pragma unroll 8
  for (int k = 0; k < 64; ++k) acc += g[gi * 64 + k] * Wc[k * 41 + o];
  out[idx] = acc;
}

// ---------------------------------------------------------------------------
extern "C" void kernel_launch(void* const* d_in, const int* in_sizes, int n_in,
                              void* d_out, int out_size, void* d_ws, size_t ws_size,
                              hipStream_t stream) {
  const float* pos   = (const float*)d_in[0];
  const int*   batch = (const int*)  d_in[1];
  const int*   face  = (const int*)  d_in[2];
  const float* W1a = (const float*)d_in[3];  const float* b1a = (const float*)d_in[4];
  const float* W1b = (const float*)d_in[5];  const float* b1b = (const float*)d_in[6];
  const float* W2a = (const float*)d_in[7];  const float* b2a = (const float*)d_in[8];
  const float* W2b = (const float*)d_in[9];  const float* b2b = (const float*)d_in[10];
  const float* W3a = (const float*)d_in[11]; const float* b3a = (const float*)d_in[12];
  const float* W3b = (const float*)d_in[13]; const float* b3b = (const float*)d_in[14];
  const float* Wc  = (const float*)d_in[15]; const float* bc  = (const float*)d_in[16];

  const int N = in_sizes[0] / 3;
  const int F = in_sizes[2] / 3;

  // workspace (16B-aligned sections):
  //   pn[N*3] f32 | bufA[N*64] f32 | g[64*64] f32 | pn16[N*3] f16 | h16[N*32] f16
  float* pn   = (float*)d_ws;
  float* bufA = pn + (size_t)N * 3;
  float* g    = bufA + (size_t)N * 64;
  _Float16* pn16 = (_Float16*)(g + 64 * 64);
  _Float16* h16  = pn16 + (size_t)N * 3 + 8;  // +8 halves keeps 16B alignment

  const int EDGE_GRID = 8192;  // single-wave blocks, grid-stride over edge tiles

  normalize_kernel<<<(N + 255) / 256, 256, 0, stream>>>(pos, pn, pn16, N);

  // layer 1: h = pn (3 features)
  zero_kernel<<<1024, 256, 0, stream>>>(bufA, N * 32);
  edge_layer_kernel<3, 32><<<EDGE_GRID, 32, 0, stream>>>(
      pn16, pn, face, W1a, b1a, W1b, b1b, bufA, N, F);
  cvt_kernel<<<1024, 256, 0, stream>>>(bufA, h16, N * 32);

  // layer 2
  zero_kernel<<<1024, 256, 0, stream>>>(bufA, N * 32);
  edge_layer_kernel<32, 32><<<EDGE_GRID, 32, 0, stream>>>(
      h16, pn, face, W2a, b2a, W2b, b2b, bufA, N, F);
  cvt_kernel<<<1024, 256, 0, stream>>>(bufA, h16, N * 32);

  // layer 3
  zero_kernel<<<1024, 256, 0, stream>>>(bufA, N * 64);
  edge_layer_kernel<32, 64><<<EDGE_GRID, 32, 0, stream>>>(
      h16, pn, face, W3a, b3a, W3b, b3b, bufA, N, F);

  // global max pool + classifier
  zero_kernel<<<64, 256, 0, stream>>>(g, 64 * 64);
  pool_kernel<<<(N + 255) / 256, 256, 0, stream>>>(bufA, batch, g, N);
  classifier_kernel<<<(64 * 41 + 127) / 128, 128, 0, stream>>>(g, Wc, bc,
                                                               (float*)d_out);
}
